// IPAdapterAttnShareProcessor_57191784513754
// MI455X (gfx1250) — matrix operand order repaired and checked by reading, not compiled, not run on hardware
//
#include <hip/hip_runtime.h>
#include <hip/hip_bf16.h>

// ---------------------------------------------------------------------------
// IP-Adapter cross-attention + temporal attention for MI455X (gfx1250).
// Large GEMMs: v_wmma_f32_16x16x32_f16 with TDM (tensor_load_to_lds) double-
// buffered LDS staging synchronized by s_wait_tensorcnt. Short-context
// attention (77 / 4 keys) and F=2 temporal attention are fp32 VALU (<4% of
// FLOPs). fp32 accumulate everywhere; f16 only as GEMM operand storage.
// ---------------------------------------------------------------------------

typedef _Float16 h16;
typedef __attribute__((ext_vector_type(16))) _Float16 v16h;
typedef __attribute__((ext_vector_type(8)))  _Float16 v8h;
typedef __attribute__((ext_vector_type(8)))  float    v8f;
typedef unsigned int u32x4 __attribute__((ext_vector_type(4)));
typedef int          i32x8 __attribute__((ext_vector_type(8)));
typedef int          i32x4 __attribute__((ext_vector_type(4)));

#if defined(__has_builtin)
#if __has_builtin(__builtin_amdgcn_tensor_load_to_lds) && __has_builtin(__builtin_amdgcn_s_wait_tensorcnt)
#define USE_TDM 1
#endif
#endif
#ifndef USE_TDM
#define USE_TDM 0
#endif

// Problem constants (from reference setup_inputs)
constexpr int F_  = 2;      // frames
constexpr int B2  = 2;      // b = B/F
constexpr int Bf  = 4;      // B
constexpr int S_  = 4096;   // spatial tokens
constexpr int C_  = 1280;   // channels
constexpr int CC  = 2048;   // context channels
constexpr int SKV = 77;     // context tokens
constexpr int T_  = 4;      // ip tokens
constexpr int R_  = 256;    // LoRA rank
constexpr int H_  = 20;     // heads
constexpr int DH  = 64;     // head dim
constexpr long Mq  = (long)F_ * B2 * S_;   // 16384 rows
constexpr int  Mkv = B2 * SKV;             // 154 rows per frame
constexpr int  LDT = 40;                   // padded LDS row stride (halfs): 32 data + 8 pad

#if USE_TDM
// Issue a TDM 2D tile load: tile_rows x 32 f16 elements from a row-major
// matrix with row stride rowStride (elements) into LDS at ldsOff, with LDS
// padding 4 dwords per 16 dwords -> 40-half padded rows. Rows >= rowsValid
// are zero-filled by TDM OOB handling.
__device__ __forceinline__ void tdm_load_2d(unsigned ldsOff, const void* gaddr,
                                            int rowStride, int rowsValid,
                                            int tileRows)
{
  const unsigned long long ga = (unsigned long long)gaddr;
  u32x4 g0;
  g0[0] = 1u;                                              // count=1 (user D#)
  g0[1] = ldsOff;                                          // lds_addr
  g0[2] = (unsigned)(ga & 0xFFFFFFFFu);                    // global_addr[31:0]
  g0[3] = (unsigned)((ga >> 32) & 0x01FFFFFFu) | (2u << 30); // addr[56:32] | type=2
  i32x8 g1;
  // data_size=1 (2B), pad_enable, pad_interval=3 (16 dwords), pad_amount=3 (4 dwords)
  g1[0] = (1 << 16) | (1 << 20) | (3 << 22) | (3 << 25);
  g1[1] = (int)(32u << 16);                                // tensor_dim0 = 32 (lo16)
  g1[2] = (int)(((unsigned)rowsValid & 0xFFFFu) << 16);    // tensor_dim1 lo16
  g1[3] = (int)((((unsigned)rowsValid >> 16) & 0xFFFFu) | (32u << 16)); // dim1 hi | tile_dim0=32
  g1[4] = (int)((unsigned)tileRows & 0xFFFFu);             // tile_dim1 (tile_dim2=0)
  g1[5] = rowStride;                                       // tensor_dim0_stride lo32
  g1[6] = 0;
  g1[7] = 0;
  const i32x4 z4 = {0, 0, 0, 0};
#if __clang_major__ >= 23
  const i32x8 z8 = {};
  __builtin_amdgcn_tensor_load_to_lds(g0, g1, z4, z4, z8, 0);
#else
  __builtin_amdgcn_tensor_load_to_lds(g0, g1, z4, z4, 0);
#endif
}
#endif

// ---------------------------------------------------------------------------
// WMMA GEMM:  out = A(MxK,f16) @ W(NxK,f16)^T (+bias), batched over z.
// Block = 256 threads = 8 waves; block tile 128x64; wave tile 16x64.
// A (128x32) and B (64x32) K-step tiles staged in LDS via TDM, double
// buffered; compute on buffer p overlaps TDM fetch of buffer p^1.
// ---------------------------------------------------------------------------
union FragH { v16h v; v8h h[2]; };

__global__ __launch_bounds__(256) void gemm_h16_wmma(
    const h16* __restrict__ A, const h16* __restrict__ W,
    const float* __restrict__ bias,
    float* __restrict__ outF, h16* __restrict__ outH,
    int M, int N, int K,
    long aStride, long wStride, long oStride)
{
  __shared__ h16 ldsA[2][128 * LDT];
  __shared__ h16 ldsB[2][64 * LDT];

  const int z = blockIdx.z;
  A += (long)z * aStride;
  W += (long)z * wStride;
  const long obase = (long)z * oStride;

  const int lane  = threadIdx.x & 31;
  const int wave  = threadIdx.x >> 5;
  const int m0blk = blockIdx.y * 128;
  const int n0    = blockIdx.x * 64;
  int mRem = M - m0blk; if (mRem < 0) mRem = 0;

  const int lm  = lane & 15;
  const int lka = (lane >> 4) << 4;   // A frag K-offset: 0 or 16 (contiguous layout)
  const int lkb = (lane >> 4) << 4;   // B frag K-offset: 0 or 16 (contiguous layout)

  auto stage = [&](int p, int k0) {
#if USE_TDM
    if (wave == 0) {
      tdm_load_2d((unsigned)(unsigned long long)&ldsA[p][0],
                  A + (long)m0blk * K + k0, K, mRem, 128);
      tdm_load_2d((unsigned)(unsigned long long)&ldsB[p][0],
                  W + (long)n0 * K + k0, K, 64, 64);
    }
#else
    for (int idx = threadIdx.x; idx < (128 * 32) / 8; idx += 256) {
      const int r = idx >> 2, cs = (idx & 3) << 3;
      v8h val = {};
      if (r < mRem) val = *(const v8h*)(A + (long)(m0blk + r) * K + k0 + cs);
      *(v8h*)&ldsA[p][r * LDT + cs] = val;
    }
    for (int idx = threadIdx.x; idx < (64 * 32) / 8; idx += 256) {
      const int r = idx >> 2, cs = (idx & 3) << 3;
      *(v8h*)&ldsB[p][r * LDT + cs] = *(const v8h*)(W + (long)(n0 + r) * K + k0 + cs);
    }
#endif
  };
  auto stage_wait = [&]() {
#if USE_TDM
    if (wave == 0) __builtin_amdgcn_s_wait_tensorcnt(0);
#endif
    __syncthreads();
  };

  stage(0, 0);

  v8f c0 = {}, c1 = {}, c2 = {}, c3 = {};
  for (int k0 = 0; k0 < K; k0 += 32) {
    const int p = (k0 >> 5) & 1;
    stage_wait();                       // buffer p ready; block synced
    if (k0 + 32 < K) stage(p ^ 1, k0 + 32);   // overlap fetch with compute

    const h16* As = &ldsA[p][(wave * 16 + lm) * LDT + lka];
    FragH a;
    a.h[0] = *(const v8h*)(As);         // K = lka .. lka+7
    a.h[1] = *(const v8h*)(As + 8);     // K = lka+8 .. lka+15

    FragH b0, b1, b2, b3;
    const h16* Bs0 = &ldsB[p][( 0 + lm) * LDT + lkb];
    const h16* Bs1 = &ldsB[p][(16 + lm) * LDT + lkb];
    const h16* Bs2 = &ldsB[p][(32 + lm) * LDT + lkb];
    const h16* Bs3 = &ldsB[p][(48 + lm) * LDT + lkb];
    b0.h[0] = *(const v8h*)Bs0; b0.h[1] = *(const v8h*)(Bs0 + 8);
    b1.h[0] = *(const v8h*)Bs1; b1.h[1] = *(const v8h*)(Bs1 + 8);
    b2.h[0] = *(const v8h*)Bs2; b2.h[1] = *(const v8h*)(Bs2 + 8);
    b3.h[0] = *(const v8h*)Bs3; b3.h[1] = *(const v8h*)(Bs3 + 8);

    c0 = __builtin_amdgcn_wmma_f32_16x16x32_f16(false, a.v, false, b0.v, (short)0, c0, false, false);
    c1 = __builtin_amdgcn_wmma_f32_16x16x32_f16(false, a.v, false, b1.v, (short)0, c1, false, false);
    c2 = __builtin_amdgcn_wmma_f32_16x16x32_f16(false, a.v, false, b2.v, (short)0, c2, false, false);
    c3 = __builtin_amdgcn_wmma_f32_16x16x32_f16(false, a.v, false, b3.v, (short)0, c3, false, false);
  }

  // C/D layout: lane<16 -> N=lane, M=r ; lane>=16 -> N=lane-16, M=8+r
  const int m0    = m0blk + wave * 16;
  const int rbase = m0 + ((lane >> 4) << 3);
  const int col0  = n0 + lm;
  v8f cc[4] = {c0, c1, c2, c3};
  #pragma unroll
  for (int j = 0; j < 4; ++j) {
    const int   col = col0 + 16 * j;
    const float bv  = bias ? bias[col] : 0.0f;
    #pragma unroll
    for (int r = 0; r < 8; ++r) {
      const int row = rbase + r;
      if (row < M) {
        const float val = cc[j][r] + bv;
        const long  o   = obase + (long)row * N + col;
        if (outF) outF[o] = val;
        if (outH) outH[o] = (h16)val;
      }
    }
  }
}

// ---------------------------------------------------------------------------
// Permute + fp32 -> f16 conversions
// ---------------------------------------------------------------------------
__global__ void k_perm_hidden(const float* __restrict__ src, h16* __restrict__ dst)
{
  const long idx = (long)blockIdx.x * 256 + threadIdx.x;
  const long tot = Mq * C_;
  if (idx >= tot) return;
  const int  c  = idx % C_;
  const long t  = idx / C_;
  const int  s  = t % S_;
  const int  fb = t / S_;          // f*b + bb
  const int  f  = fb / B2, bb = fb % B2;
  dst[idx] = (h16)src[(((long)(bb * F_ + f) * S_) + s) * C_ + c];
}

__global__ void k_perm_ctx(const float* __restrict__ src, h16* __restrict__ dst)
{
  const long idx = (long)blockIdx.x * 256 + threadIdx.x;
  const long tot = (long)F_ * B2 * SKV * CC;
  if (idx >= tot) return;
  const int  c  = idx % CC;
  const long t  = idx / CC;
  const int  s  = t % SKV;
  const int  fb = t / SKV;
  const int  f  = fb / B2, bb = fb % B2;
  dst[idx] = (h16)src[(((long)(bb * F_ + f) * SKV) + s) * CC + c];
}

__global__ void k_f2h(const float* __restrict__ src, h16* __restrict__ dst, long n)
{
  const long i = (long)blockIdx.x * 256 + threadIdx.x;
  if (i < n) dst[i] = (h16)src[i];
}

// merged[f,o,i] = W[o,i] + sum_r up[f,o,r]*down[f,r,i]  -> f16
__global__ void k_merge_lora(const float* __restrict__ W, const float* __restrict__ up,
                             const float* __restrict__ dn, h16* __restrict__ out,
                             int O, int I)
{
  const long idx = (long)blockIdx.x * 256 + threadIdx.x;
  const long tot = (long)F_ * O * I;
  if (idx >= tot) return;
  const int  i = idx % I;
  const long t = idx / I;
  const int  o = t % O;
  const int  f = t / O;
  float acc = W[(long)o * I + i];
  const float* uprow = up + ((long)f * O + o) * R_;
  const float* dncol = dn + (long)f * R_ * I + i;
  #pragma unroll 4
  for (int r = 0; r < R_; ++r) acc += uprow[r] * dncol[(long)r * I];
  out[idx] = (h16)acc;
}

// ip k/v projections (tiny: b*T=8 rows)
__global__ void k_ipkv(const float* __restrict__ iph, const float* __restrict__ Wk,
                       const float* __restrict__ Wv, float* __restrict__ ipk,
                       float* __restrict__ ipv)
{
  const int idx = blockIdx.x * 256 + threadIdx.x;   // over b*T*C
  if (idx >= B2 * T_ * C_) return;
  const int o = idx % C_, row = idx / C_;
  const float* x  = iph + (long)row * CC;
  const float* wk = Wk + (long)o * CC;
  const float* wv = Wv + (long)o * CC;
  float ak = 0.f, av = 0.f;
  for (int c = 0; c < CC; ++c) { const float xv = x[c]; ak += xv * wk[c]; av += xv * wv[c]; }
  ipk[idx] = ak; ipv[idx] = av;
}

// ---------------------------------------------------------------------------
// Fused cross-attention (77 keys) + IP attention (4 keys), separate softmaxes.
// Block = 64 threads = 64 queries of one (f,b,h). K/V/ipK/ipV + Q/score
// staging in LDS (~62 KB).
// ---------------------------------------------------------------------------
__global__ __launch_bounds__(64) void k_attn(
    const float* __restrict__ q, const float* __restrict__ k,
    const float* __restrict__ v, const float* __restrict__ ipk,
    const float* __restrict__ ipv, h16* __restrict__ outH)
{
  __shared__ float kk[SKV][DH];
  __shared__ float vv[SKV][DH];
  __shared__ float ik[T_][DH];
  __shared__ float iv[T_][DH];
  __shared__ float uni[64 * 81];   // Q staging (stride 65) -> scores (stride 81) -> out staging

  const int fb  = blockIdx.z;      // f*b + bb
  const int h   = blockIdx.y;
  const int s0  = blockIdx.x * 64;
  const int tid = threadIdx.x;
  const int bb  = fb % B2;

  for (int idx = tid; idx < SKV * DH; idx += 64) {
    const int t = idx / DH, d = idx % DH;
    kk[t][d] = k[((long)fb * SKV + t) * C_ + h * DH + d];
    vv[t][d] = v[((long)fb * SKV + t) * C_ + h * DH + d];
  }
  for (int idx = tid; idx < T_ * DH; idx += 64) {
    const int t = idx / DH, d = idx % DH;
    ik[t][d] = ipk[((long)bb * T_ + t) * C_ + h * DH + d];
    iv[t][d] = ipv[((long)bb * T_ + t) * C_ + h * DH + d];
  }
  for (int idx = tid; idx < 64 * DH; idx += 64) {
    const int t = idx / DH, d = idx % DH;
    uni[t * 65 + d] = q[((long)fb * S_ + s0 + t) * C_ + h * DH + d];
  }
  __syncthreads();

  float qr[DH];
  #pragma unroll
  for (int d = 0; d < DH; ++d) qr[d] = uni[tid * 65 + d];
  __syncthreads();

  float* sc = uni;                 // reuse as scores, stride 81 (conflict-free)
  const float scale = 0.125f;      // Dh^-0.5
  float mx = -3.0e38f;
  for (int t = 0; t < SKV; ++t) {
    float s = 0.f;
    #pragma unroll
    for (int d = 0; d < DH; ++d) s += qr[d] * kk[t][d];
    s *= scale;
    sc[tid * 81 + t] = s;
    mx = fmaxf(mx, s);
  }
  float sum = 0.f;
  for (int t = 0; t < SKV; ++t) {
    const float e = __expf(sc[tid * 81 + t] - mx);
    sc[tid * 81 + t] = e; sum += e;
  }
  const float inv = 1.0f / sum;

  float acc[DH];
  #pragma unroll
  for (int d = 0; d < DH; ++d) acc[d] = 0.f;
  for (int t = 0; t < SKV; ++t) {
    const float w = sc[tid * 81 + t] * inv;
    #pragma unroll
    for (int d = 0; d < DH; ++d) acc[d] += w * vv[t][d];
  }

  // IP attention: its own softmax over 4 keys, added with scale 1.0
  float is[T_], imx = -3.0e38f;
  #pragma unroll
  for (int t = 0; t < T_; ++t) {
    float s = 0.f;
    #pragma unroll
    for (int d = 0; d < DH; ++d) s += qr[d] * ik[t][d];
    is[t] = s * scale;
    imx = fmaxf(imx, is[t]);
  }
  float isum = 0.f;
  #pragma unroll
  for (int t = 0; t < T_; ++t) { is[t] = __expf(is[t] - imx); isum += is[t]; }
  const float iinv = 1.0f / isum;
  #pragma unroll
  for (int t = 0; t < T_; ++t) {
    const float w = is[t] * iinv;
    #pragma unroll
    for (int d = 0; d < DH; ++d) acc[d] += w * iv[t][d];
  }

  __syncthreads();
  #pragma unroll
  for (int d = 0; d < DH; ++d) uni[tid * 65 + d] = acc[d];
  __syncthreads();
  for (int idx = tid; idx < 64 * DH; idx += 64) {
    const int t = idx / DH, d = idx % DH;
    outH[((long)fb * S_ + s0 + t) * C_ + h * DH + d] = (h16)uni[t * 65 + d];
  }
}

// ---------------------------------------------------------------------------
// LayerNorm with (f,b,s) -> (n=b*S+s, f) permutation folded in; f16 out.
// ---------------------------------------------------------------------------
__global__ __launch_bounds__(256) void k_ln(
    const float* __restrict__ y, const float* __restrict__ g,
    const float* __restrict__ be, h16* __restrict__ outH)
{
  __shared__ float s1[256], s2[256];
  const long xr = blockIdx.x;            // 0..Mq-1  (row of x = n*F + f)
  const int  f  = (int)(xr % F_);
  const long n  = xr / F_;
  const int  bb = (int)(n / S_);
  const int  s  = (int)(n % S_);
  const float* row = y + (((long)(f * B2 + bb) * S_) + s) * C_;

  float a = 0.f, b2 = 0.f;
  for (int c = threadIdx.x; c < C_; c += 256) { const float xv = row[c]; a += xv; b2 += xv * xv; }
  s1[threadIdx.x] = a; s2[threadIdx.x] = b2;
  __syncthreads();
  for (int st = 128; st > 0; st >>= 1) {
    if (threadIdx.x < st) { s1[threadIdx.x] += s1[threadIdx.x + st]; s2[threadIdx.x] += s2[threadIdx.x + st]; }
    __syncthreads();
  }
  const float mu  = s1[0] * (1.0f / C_);
  const float var = s2[0] * (1.0f / C_) - mu * mu;
  const float rs  = rsqrtf(var + 1e-6f);
  for (int c = threadIdx.x; c < C_; c += 256)
    outH[xr * C_ + c] = (h16)((row[c] - mu) * rs * g[c] + be[c]);
}

// ---------------------------------------------------------------------------
// Temporal attention over F=2 frames per n row: 4 dots of length C, 2 softmaxes.
// ---------------------------------------------------------------------------
__global__ __launch_bounds__(64) void k_tattn(
    const float* __restrict__ tq, const float* __restrict__ tk,
    const float* __restrict__ tv, h16* __restrict__ outH)
{
  __shared__ float red[4][64];
  const long n = blockIdx.x;
  const float* q0 = tq + (n * 2 + 0) * C_;
  const float* q1 = tq + (n * 2 + 1) * C_;
  const float* k0 = tk + (n * 2 + 0) * C_;
  const float* k1 = tk + (n * 2 + 1) * C_;
  float p00 = 0, p01 = 0, p10 = 0, p11 = 0;
  for (int c = threadIdx.x; c < C_; c += 64) {
    const float a0 = q0[c], a1 = q1[c], x0 = k0[c], x1 = k1[c];
    p00 += a0 * x0; p01 += a0 * x1; p10 += a1 * x0; p11 += a1 * x1;
  }
  red[0][threadIdx.x] = p00; red[1][threadIdx.x] = p01;
  red[2][threadIdx.x] = p10; red[3][threadIdx.x] = p11;
  __syncthreads();
  for (int st = 32; st > 0; st >>= 1) {
    if (threadIdx.x < st)
      for (int p = 0; p < 4; ++p) red[p][threadIdx.x] += red[p][threadIdx.x + st];
    __syncthreads();
  }
  const float scl = rsqrtf((float)C_);   // C^-0.5 for temporal attention
  const float s00 = red[0][0] * scl, s01 = red[1][0] * scl;
  const float s10 = red[2][0] * scl, s11 = red[3][0] * scl;
  const float m0 = fmaxf(s00, s01), m1 = fmaxf(s10, s11);
  const float e00 = __expf(s00 - m0), e01 = __expf(s01 - m0);
  const float e10 = __expf(s10 - m1), e11 = __expf(s11 - m1);
  const float i0 = 1.0f / (e00 + e01), i1 = 1.0f / (e10 + e11);
  const float w00 = e00 * i0, w01 = e01 * i0, w10 = e10 * i1, w11 = e11 * i1;
  const float* v0 = tv + (n * 2 + 0) * C_;
  const float* v1 = tv + (n * 2 + 1) * C_;
  for (int c = threadIdx.x; c < C_; c += 64) {
    const float a = v0[c], bvv = v1[c];
    outH[(n * 2 + 0) * C_ + c] = (h16)(w00 * a + w01 * bvv);
    outH[(n * 2 + 1) * C_ + c] = (h16)(w10 * a + w11 * bvv);
  }
}

// final: out[(bb*F+f)*S+s, c] = y[(f*b+bb)*S+s, c] + tproj[(bb*S+s)*F+f, c]
__global__ void k_final(const float* __restrict__ y, const float* __restrict__ tproj,
                        float* __restrict__ out)
{
  const long idx = (long)blockIdx.x * 256 + threadIdx.x;
  const long tot = (long)Bf * S_ * C_;
  if (idx >= tot) return;
  const int  c  = idx % C_;
  const long r  = idx / C_;
  const int  s  = r % S_;
  const int  bf = r / S_;
  const int  f  = bf % F_, bb = bf / F_;
  out[idx] = y[(((long)(f * B2 + bb) * S_) + s) * C_ + c]
           + tproj[(((long)(bb * (long)S_ + s) * F_) + f) * C_ + c];
}

// ---------------------------------------------------------------------------
extern "C" void kernel_launch(void* const* d_in, const int* in_sizes, int n_in,
                              void* d_out, int out_size, void* d_ws, size_t ws_size,
                              hipStream_t stream) {
  (void)in_sizes; (void)n_in; (void)out_size; (void)ws_size;
  const float* hidden = (const float*)d_in[0];
  const float* enc    = (const float*)d_in[1];
  const float* iph    = (const float*)d_in[2];
  const float* Wq     = (const float*)d_in[3];
  const float* Wk     = (const float*)d_in[4];
  const float* Wv     = (const float*)d_in[5];
  const float* Wo     = (const float*)d_in[6];
  const float* bo     = (const float*)d_in[7];
  const float* lq_up  = (const float*)d_in[8];
  const float* lq_dn  = (const float*)d_in[9];
  const float* lk_up  = (const float*)d_in[10];
  const float* lk_dn  = (const float*)d_in[11];
  const float* lv_up  = (const float*)d_in[12];
  const float* lv_dn  = (const float*)d_in[13];
  const float* lo_up  = (const float*)d_in[14];
  const float* lo_dn  = (const float*)d_in[15];
  const float* Wk_ip  = (const float*)d_in[16];
  const float* Wv_ip  = (const float*)d_in[17];
  const float* Wti    = (const float*)d_in[18];
  const float* bti    = (const float*)d_in[19];
  const float* ln_g   = (const float*)d_in[20];
  const float* ln_b   = (const float*)d_in[21];
  const float* Wtq    = (const float*)d_in[22];
  const float* btq    = (const float*)d_in[23];
  const float* Wtk    = (const float*)d_in[24];
  const float* btk    = (const float*)d_in[25];
  const float* Wtv    = (const float*)d_in[26];
  const float* btv    = (const float*)d_in[27];
  const float* Wto    = (const float*)d_in[28];
  const float* bto    = (const float*)d_in[29];
  float* out = (float*)d_out;

  // workspace bump allocator (256B aligned)
  char*  ws  = (char*)d_ws;
  size_t off = 0;
  auto alloc = [&](size_t bytes) -> char* {
    size_t o = (off + 255) & ~(size_t)255;
    off = o + bytes;
    return ws + o;
  };
  h16*   Xq_h   = (h16*)alloc(Mq * C_ * 2);                 // reused later as i_h
  h16*   Cx_h   = (h16*)alloc((long)F_ * Mkv * CC * 2);
  h16*   Wq_m   = (h16*)alloc((long)F_ * C_ * C_ * 2);
  h16*   Wk_m   = (h16*)alloc((long)F_ * C_ * CC * 2);
  h16*   Wv_m   = (h16*)alloc((long)F_ * C_ * CC * 2);
  h16*   Wo_m   = (h16*)alloc((long)F_ * C_ * C_ * 2);
  h16*   Wti_h  = (h16*)alloc((long)C_ * C_ * 2);
  h16*   Wtq_h  = (h16*)alloc((long)C_ * C_ * 2);
  h16*   Wtk_h  = (h16*)alloc((long)C_ * C_ * 2);
  h16*   Wtv_h  = (h16*)alloc((long)C_ * C_ * 2);
  h16*   Wto_h  = (h16*)alloc((long)C_ * C_ * 2);
  float* q_f    = (float*)alloc(Mq * C_ * 4);               // reused later as tq
  float* k_f    = (float*)alloc((long)F_ * Mkv * C_ * 4);
  float* v_f    = (float*)alloc((long)F_ * Mkv * C_ * 4);
  float* ipk_f  = (float*)alloc((long)B2 * T_ * C_ * 4);
  float* ipv_f  = (float*)alloc((long)B2 * T_ * C_ * 4);
  h16*   attn_h = (h16*)alloc(Mq * C_ * 2);                 // reused later as tatt
  float* y_f    = (float*)alloc(Mq * C_ * 4);
  h16*   n_h    = (h16*)alloc(Mq * C_ * 2);
  float* tk_f   = (float*)alloc(Mq * C_ * 4);
  float* tv_f   = (float*)alloc(Mq * C_ * 4);
  float* tpj_f  = (float*)alloc(Mq * C_ * 4);

  const dim3 blk(256);

  // 1) permute + f32->f16 inputs
  { long tot = Mq * C_;              k_perm_hidden<<<dim3((tot + 255) / 256), blk, 0, stream>>>(hidden, Xq_h); }
  { long tot = (long)F_ * Mkv * CC;  k_perm_ctx<<<dim3((tot + 255) / 256), blk, 0, stream>>>(enc, Cx_h); }

  // 2) LoRA weight merges (f16 out) + temporal weight conversions
  { long tot = (long)F_ * C_ * C_;  k_merge_lora<<<dim3((tot + 255) / 256), blk, 0, stream>>>(Wq, lq_up, lq_dn, Wq_m, C_, C_); }
  { long tot = (long)F_ * C_ * CC;  k_merge_lora<<<dim3((tot + 255) / 256), blk, 0, stream>>>(Wk, lk_up, lk_dn, Wk_m, C_, CC); }
  { long tot = (long)F_ * C_ * CC;  k_merge_lora<<<dim3((tot + 255) / 256), blk, 0, stream>>>(Wv, lv_up, lv_dn, Wv_m, C_, CC); }
  { long tot = (long)F_ * C_ * C_;  k_merge_lora<<<dim3((tot + 255) / 256), blk, 0, stream>>>(Wo, lo_up, lo_dn, Wo_m, C_, C_); }
  { long n = (long)C_ * C_; dim3 g((n + 255) / 256);
    k_f2h<<<g, blk, 0, stream>>>(Wti, Wti_h, n);
    k_f2h<<<g, blk, 0, stream>>>(Wtq, Wtq_h, n);
    k_f2h<<<g, blk, 0, stream>>>(Wtk, Wtk_h, n);
    k_f2h<<<g, blk, 0, stream>>>(Wtv, Wtv_h, n);
    k_f2h<<<g, blk, 0, stream>>>(Wto, Wto_h, n); }

  // 3) Q projection (per-frame merged weights), WMMA
  {
    dim3 g(C_ / 64, (B2 * S_ + 127) / 128, F_);
    gemm_h16_wmma<<<g, blk, 0, stream>>>(Xq_h, Wq_m, nullptr, q_f, nullptr,
        B2 * S_, C_, C_, (long)B2 * S_ * C_, (long)C_ * C_, (long)B2 * S_ * C_);
  }
  // 4) K / V projections (154 rows per frame), WMMA
  {
    dim3 g(C_ / 64, (Mkv + 127) / 128, F_);
    gemm_h16_wmma<<<g, blk, 0, stream>>>(Cx_h, Wk_m, nullptr, k_f, nullptr,
        Mkv, C_, CC, (long)Mkv * CC, (long)C_ * CC, (long)Mkv * C_);
    gemm_h16_wmma<<<g, blk, 0, stream>>>(Cx_h, Wv_m, nullptr, v_f, nullptr,
        Mkv, C_, CC, (long)Mkv * CC, (long)C_ * CC, (long)Mkv * C_);
  }
  // 5) IP k/v projections (tiny)
  { int tot = B2 * T_ * C_; k_ipkv<<<dim3((tot + 255) / 256), blk, 0, stream>>>(iph, Wk_ip, Wv_ip, ipk_f, ipv_f); }

  // 6) fused cross + IP attention -> f16
  {
    dim3 g(S_ / 64, H_, F_ * B2);
    k_attn<<<g, dim3(64), 0, stream>>>(q_f, k_f, v_f, ipk_f, ipv_f, attn_h);
  }
  // 7) output projection with per-frame merged Wo + bo
  {
    dim3 g(C_ / 64, (B2 * S_ + 127) / 128, F_);
    gemm_h16_wmma<<<g, blk, 0, stream>>>(attn_h, Wo_m, bo, y_f, nullptr,
        B2 * S_, C_, C_, (long)B2 * S_ * C_, (long)C_ * C_, (long)B2 * S_ * C_);
  }
  // 8) temporal: LN (+permute) -> i -> tq/tk/tv -> F=2 attention -> Wto -> residual
  k_ln<<<dim3((unsigned)Mq), blk, 0, stream>>>(y_f, ln_g, ln_b, n_h);

  h16* i_h = Xq_h;          // Xq dead after step 3
  {
    dim3 g(C_ / 64, (unsigned)(Mq / 128), 1);
    gemm_h16_wmma<<<g, blk, 0, stream>>>(n_h, Wti_h, bti, nullptr, i_h,
        (int)Mq, C_, C_, 0, 0, 0);
    float* tq_f = q_f;      // q dead after step 6
    gemm_h16_wmma<<<g, blk, 0, stream>>>(i_h, Wtq_h, btq, tq_f, nullptr,
        (int)Mq, C_, C_, 0, 0, 0);
    gemm_h16_wmma<<<g, blk, 0, stream>>>(i_h, Wtk_h, btk, tk_f, nullptr,
        (int)Mq, C_, C_, 0, 0, 0);
    gemm_h16_wmma<<<g, blk, 0, stream>>>(i_h, Wtv_h, btv, tv_f, nullptr,
        (int)Mq, C_, C_, 0, 0, 0);

    h16* tatt_h = attn_h;   // attn dead after step 7
    k_tattn<<<dim3((unsigned)(Mq / F_)), dim3(64), 0, stream>>>(tq_f, tk_f, tv_f, tatt_h);

    gemm_h16_wmma<<<g, blk, 0, stream>>>(tatt_h, Wto_h, bto, tpj_f, nullptr,
        (int)Mq, C_, C_, 0, 0, 0);
  }
  { long tot = (long)Bf * S_ * C_; k_final<<<dim3((tot + 255) / 256), blk, 0, stream>>>(y_f, tpj_f, out); }
}
